// GIN0_9131100472083
// MI455X (gfx1250) — matrix operand order repaired and checked by reading, not compiled, not run on hardware
//
#include <hip/hip_runtime.h>
#include <hip/hip_bf16.h>

typedef __attribute__((ext_vector_type(16))) _Float16 v16h;
typedef __attribute__((ext_vector_type(8)))  _Float16 v8h;
typedef __attribute__((ext_vector_type(2)))  _Float16 v2h;
typedef __attribute__((ext_vector_type(8)))  float    v8f;

#define HID 128
#define WT_STRIDE 136   // padded f16 stride for W^T in LDS (272B rows, conflict-free b128)
#define TS 136          // padded f16 stride for t tile in LDS
#define RPB 64          // rows per block for GEMM kernels
#define MT 4            // m-tiles per wave (RPB/16)
#define BN_EPS 1e-5f

// ---------------------------------------------------------------------------
// Packed f16 atomic add (global_atomic_pk_add_f16), with CAS fallback.
// ---------------------------------------------------------------------------
__device__ inline void atomic_pk_add_f16(_Float16* addr, v2h val) {
#if __has_builtin(__builtin_amdgcn_global_atomic_fadd_v2f16)
    __builtin_amdgcn_global_atomic_fadd_v2f16(
        (__attribute__((address_space(1))) v2h*)addr, val);
#else
    unsigned int* ua = (unsigned int*)addr;
    unsigned int old = *ua, assumed;
    do {
        assumed = old;
        v2h cur;
        __builtin_memcpy(&cur, &assumed, 4);
        cur = cur + val;
        unsigned int nv;
        __builtin_memcpy(&nv, &cur, 4);
        old = atomicCAS(ua, assumed, nv);
    } while (old != assumed);
#endif
}

// ---------------------------------------------------------------------------
// Zero fill (dword granularity; all f16 buffers have even element counts)
// ---------------------------------------------------------------------------
__global__ void gin_zero(float* __restrict__ p, long long n) {
    long long i = (long long)blockIdx.x * blockDim.x + threadIdx.x;
    if (i < n) p[i] = 0.0f;
}

// ---------------------------------------------------------------------------
// Layer-1 edge aggregation over raw 2-feature f32 input: agg2[dst] += x[src]
// ---------------------------------------------------------------------------
__global__ void gin_agg2(const float* __restrict__ x, const int* __restrict__ src,
                         const int* __restrict__ dst, float* __restrict__ agg2, int E) {
    int e = blockIdx.x * blockDim.x + threadIdx.x;
    if (e < E) {
        int s = src[e], d = dst[e];
        float2 v = *(const float2*)(x + (size_t)s * 2);
        atomicAdd(agg2 + (size_t)d * 2 + 0, v.x);
        atomicAdd(agg2 + (size_t)d * 2 + 1, v.y);
    }
}

// ---------------------------------------------------------------------------
// 128-feature f16 edge aggregation: 32 lanes/edge, 8B gather + 2 pk-f16 atomics.
// h fits in L2 (12.8MB << 192MB): gathers + atomics resolve in L2.
// ---------------------------------------------------------------------------
__global__ void gin_agg128(const _Float16* __restrict__ h, const int* __restrict__ src,
                           const int* __restrict__ dst, _Float16* __restrict__ agg, int E) {
    long long tid = (long long)blockIdx.x * blockDim.x + threadIdx.x;
    int e = (int)(tid >> 5);
    int c = ((int)(tid & 31)) << 2;     // 4 halves per lane
    if (e < E) {
        int s = src[e], d = dst[e];
        const v2h* ph = (const v2h*)(h + (size_t)s * HID + c);
        v2h a0 = ph[0], a1 = ph[1];
        _Float16* pd = agg + (size_t)d * HID + c;
        atomic_pk_add_f16(pd + 0, a0);
        atomic_pk_add_f16(pd + 2, a1);
    }
}

// ---------------------------------------------------------------------------
// Layer-1 first linear (in-dim 2): t = relu((x+agg2) @ W1 + b1)  -> f16
// ---------------------------------------------------------------------------
__global__ void gin_mlp1(const float* __restrict__ x, const float* __restrict__ agg2,
                         const float* __restrict__ W1, const float* __restrict__ b1,
                         _Float16* __restrict__ out, int N) {
    long long tid = (long long)blockIdx.x * blockDim.x + threadIdx.x;
    int n = (int)(tid >> 7);
    int j = (int)(tid & 127);
    if (n < N) {
        float s0 = x[(size_t)n * 2 + 0] + agg2[(size_t)n * 2 + 0];
        float s1 = x[(size_t)n * 2 + 1] + agg2[(size_t)n * 2 + 1];
        float z = fmaf(s0, W1[j], fmaf(s1, W1[HID + j], b1[j]));
        out[(size_t)n * HID + j] = (_Float16)fmaxf(z, 0.0f);
    }
}

// ---------------------------------------------------------------------------
// Helpers shared by the WMMA kernels
// ---------------------------------------------------------------------------
__device__ inline void stage_wt_f16(const float* __restrict__ W, _Float16* __restrict__ sWt) {
    // W row-major [K][N] f32  ->  sWt[n*WT_STRIDE + k] f16 (transposed)
    for (int i4 = threadIdx.x * 4; i4 < HID * HID; i4 += blockDim.x * 4) {
        int k = i4 >> 7;
        int n = i4 & 127;
        float4 w = *(const float4*)(W + i4);
        sWt[(n + 0) * WT_STRIDE + k] = (_Float16)w.x;
        sWt[(n + 1) * WT_STRIDE + k] = (_Float16)w.y;
        sWt[(n + 2) * WT_STRIDE + k] = (_Float16)w.z;
        sWt[(n + 3) * WT_STRIDE + k] = (_Float16)w.w;
    }
}

__device__ inline v16h cat8(v8h lo, v8h hi) {
    return __builtin_shufflevector(lo, hi, 0, 1, 2, 3, 4, 5, 6, 7,
                                   8, 9, 10, 11, 12, 13, 14, 15);
}

// ---------------------------------------------------------------------------
// Fused GIN MLP (layers 2..L), all-f16 activations, in-place on H:
//   t  = relu((H + AGG) @ W1 + b1)     (t kept in LDS, f16)
//   H  = BN(relu(t @ W2 + b2))
// Block: 256 threads = 8 wave32; tile = 64 rows x 128 cols.
// Wave w owns col tile [16w,16w+16); 4 m-tiles share each B-fragment.
// ---------------------------------------------------------------------------
__global__ __launch_bounds__(256)
void gin_fused_mlp(const _Float16* __restrict__ H, const _Float16* __restrict__ AGG,
                   const float* __restrict__ W1, const float* __restrict__ b1,
                   const float* __restrict__ W2, const float* __restrict__ b2,
                   const float* __restrict__ gamma, const float* __restrict__ beta,
                   const float* __restrict__ mean, const float* __restrict__ var,
                   _Float16* __restrict__ Hout, int nrows) {
    __shared__ _Float16 sW1[HID * WT_STRIDE];
    __shared__ _Float16 sW2[HID * WT_STRIDE];
    __shared__ _Float16 sT[RPB * TS];

    const int lane    = threadIdx.x & 31;
    const int wave    = threadIdx.x >> 5;
    const int colBase = wave * 16;
    const int rowBase = blockIdx.x * RPB;
    const int l16     = lane & 15;
    const int hi      = lane >> 4;
    const int n       = colBase + l16;

    // A-row f16 element offsets per m-tile (clamped; last block only)
    size_t aoff[MT];
#pragma unroll
    for (int mt = 0; mt < MT; ++mt) {
        int m = rowBase + mt * 16 + l16;
        if (m > nrows - 1) m = nrows - 1;
        aoff[mt] = (size_t)m * HID + hi * 8;
        // Warm L2/L0 with this block's A/AGG rows while we stage weights.
        __builtin_prefetch(H + aoff[mt], 0, 3);
        __builtin_prefetch(AGG + aoff[mt], 0, 3);
    }

    stage_wt_f16(W1, sW1);
    stage_wt_f16(W2, sW2);
    __syncthreads();

    // ---------------- phase 1: acc = (H+AGG) @ W1 ----------------
    v8f acc[MT];
#pragma unroll
    for (int mt = 0; mt < MT; ++mt) acc[mt] = (v8f){0.f, 0.f, 0.f, 0.f, 0.f, 0.f, 0.f, 0.f};

    const _Float16* pB1 = &sW1[n * WT_STRIDE + hi * 8];
#pragma unroll
    for (int kk = 0; kk < HID; kk += 32) {
        v16h b = cat8(*(const v8h*)(pB1 + kk), *(const v8h*)(pB1 + kk + 16));
#pragma unroll
        for (int mt = 0; mt < MT; ++mt) {
            v8h h0 = *(const v8h*)(H + aoff[mt] + kk);        // K = kk+8*hi+{0..7}
            v8h h1 = *(const v8h*)(H + aoff[mt] + kk + 16);   // K = kk+16+8*hi+{0..7}
            v8h g0 = *(const v8h*)(AGG + aoff[mt] + kk);
            v8h g1 = *(const v8h*)(AGG + aoff[mt] + kk + 16);
            h0 = h0 + g0;                                     // v_pk_add_f16
            h1 = h1 + g1;
            acc[mt] = __builtin_amdgcn_wmma_f32_16x16x32_f16(
                false, cat8(h0, h1), false, b, (short)0, acc[mt], false, false);
        }
    }

    // t = relu(acc + b1) -> LDS (f16)
    {
        float bi1 = b1[n];
#pragma unroll
        for (int mt = 0; mt < MT; ++mt)
#pragma unroll
            for (int v = 0; v < 8; ++v) {
                int ml = mt * 16 + v + hi * 8;          // C layout: VGPR v -> M = v + 8*hi
                sT[ml * TS + n] = (_Float16)fmaxf(acc[mt][v] + bi1, 0.0f);
            }
    }
    __syncthreads();

    // ---------------- phase 2: acc = t @ W2 ----------------
#pragma unroll
    for (int mt = 0; mt < MT; ++mt) acc[mt] = (v8f){0.f, 0.f, 0.f, 0.f, 0.f, 0.f, 0.f, 0.f};

    const _Float16* pB2 = &sW2[n * WT_STRIDE + hi * 8];
#pragma unroll
    for (int kk = 0; kk < HID; kk += 32) {
        v16h b = cat8(*(const v8h*)(pB2 + kk), *(const v8h*)(pB2 + kk + 16));
#pragma unroll
        for (int mt = 0; mt < MT; ++mt) {
            const _Float16* pa = &sT[(mt * 16 + l16) * TS + hi * 8];
            v16h a = cat8(*(const v8h*)(pa + kk), *(const v8h*)(pa + kk + 16));
            acc[mt] = __builtin_amdgcn_wmma_f32_16x16x32_f16(
                false, a, false, b, (short)0, acc[mt], false, false);
        }
    }

    // epilogue: bias + ReLU + inference-BN affine -> f16
    float bi2 = b2[n];
    float sc = gamma[n] * rsqrtf(var[n] + BN_EPS);
    float mu = mean[n];
    float bb = beta[n];
#pragma unroll
    for (int mt = 0; mt < MT; ++mt)
#pragma unroll
        for (int v = 0; v < 8; ++v) {
            int mm = rowBase + mt * 16 + v + hi * 8;
            float z = fmaxf(acc[mt][v] + bi2, 0.0f);
            z = (z - mu) * sc + bb;
            if (mm < nrows) Hout[(size_t)mm * HID + n] = (_Float16)z;
        }
}

// ---------------------------------------------------------------------------
// Single WMMA GEMM: Y = epilogue(X @ W + bias), X/Y f16, W f32 (staged to LDS).
// Same 64x128 block tiling. DO_BN selects ReLU+BN vs plain ReLU epilogue.
// ---------------------------------------------------------------------------
template <bool DO_BN>
__global__ __launch_bounds__(256)
void gin_gemm_single(const _Float16* __restrict__ X, const float* __restrict__ W,
                     const float* __restrict__ bias,
                     const float* __restrict__ gamma, const float* __restrict__ beta,
                     const float* __restrict__ mean, const float* __restrict__ var,
                     _Float16* __restrict__ Y, int nrows) {
    __shared__ _Float16 sWt[HID * WT_STRIDE];

    const int lane    = threadIdx.x & 31;
    const int wave    = threadIdx.x >> 5;
    const int colBase = wave * 16;
    const int rowBase = blockIdx.x * RPB;
    const int l16     = lane & 15;
    const int hi      = lane >> 4;
    const int n       = colBase + l16;

    size_t aoff[MT];
#pragma unroll
    for (int mt = 0; mt < MT; ++mt) {
        int m = rowBase + mt * 16 + l16;
        if (m > nrows - 1) m = nrows - 1;
        aoff[mt] = (size_t)m * HID + hi * 8;
        __builtin_prefetch(X + aoff[mt], 0, 3);
    }

    stage_wt_f16(W, sWt);
    __syncthreads();

    v8f acc[MT];
#pragma unroll
    for (int mt = 0; mt < MT; ++mt) acc[mt] = (v8f){0.f, 0.f, 0.f, 0.f, 0.f, 0.f, 0.f, 0.f};

    const _Float16* pB = &sWt[n * WT_STRIDE + hi * 8];
#pragma unroll
    for (int kk = 0; kk < HID; kk += 32) {
        v16h b = cat8(*(const v8h*)(pB + kk), *(const v8h*)(pB + kk + 16));
#pragma unroll
        for (int mt = 0; mt < MT; ++mt) {
            v16h a = cat8(*(const v8h*)(X + aoff[mt] + kk),
                          *(const v8h*)(X + aoff[mt] + kk + 16));
            acc[mt] = __builtin_amdgcn_wmma_f32_16x16x32_f16(
                false, a, false, b, (short)0, acc[mt], false, false);
        }
    }

    float bi = bias[n];
    float sc = 1.0f, mu = 0.0f, bb = 0.0f;
    if (DO_BN) {
        sc = gamma[n] * rsqrtf(var[n] + BN_EPS);
        mu = mean[n];
        bb = beta[n];
    }
#pragma unroll
    for (int mt = 0; mt < MT; ++mt)
#pragma unroll
        for (int v = 0; v < 8; ++v) {
            int mm = rowBase + mt * 16 + v + hi * 8;
            float z = fmaxf(acc[mt][v] + bi, 0.0f);
            if (DO_BN) z = (z - mu) * sc + bb;
            if (mm < nrows) Y[(size_t)mm * HID + n] = (_Float16)z;
        }
}

// ---------------------------------------------------------------------------
// Global add-pool (f16): pooled[batch[n]] += h[n], pk-f16 atomics
// ---------------------------------------------------------------------------
__global__ void gin_pool(const _Float16* __restrict__ h, const int* __restrict__ batch,
                         _Float16* __restrict__ pooled, int N) {
    long long tid = (long long)blockIdx.x * blockDim.x + threadIdx.x;
    int nd = (int)(tid >> 5);
    int c  = ((int)(tid & 31)) << 2;
    if (nd < N) {
        int g = batch[nd];
        const v2h* ph = (const v2h*)(h + (size_t)nd * HID + c);
        v2h a0 = ph[0], a1 = ph[1];
        _Float16* pd = pooled + (size_t)g * HID + c;
        atomic_pk_add_f16(pd + 0, a0);
        atomic_pk_add_f16(pd + 2, a1);
    }
}

// ---------------------------------------------------------------------------
// Final mat-vec: out[g] = q[g,:] . w + b   (one wave32 per graph, f32 out)
// ---------------------------------------------------------------------------
__global__ void gin_lin2(const _Float16* __restrict__ q, const float* __restrict__ w,
                         const float* __restrict__ b, float* __restrict__ out, int G) {
    int gid  = (blockIdx.x * blockDim.x + threadIdx.x) >> 5;
    int lane = threadIdx.x & 31;
    if (gid < G) {
        float s = 0.0f;
#pragma unroll
        for (int j = lane; j < HID; j += 32)
            s = fmaf((float)q[(size_t)gid * HID + j], w[j], s);
        for (int o = 16; o > 0; o >>= 1) s += __shfl_down(s, o, 32);
        if (lane == 0) out[gid] = s + b[0];
    }
}

// ---------------------------------------------------------------------------
extern "C" void kernel_launch(void* const* d_in, const int* in_sizes, int n_in,
                              void* d_out, int out_size, void* d_ws, size_t ws_size,
                              hipStream_t stream) {
    const float* x    = (const float*)d_in[0];
    const int*   ei   = (const int*)d_in[1];
    const int*   bat  = (const int*)d_in[2];
    const float* c1W1 = (const float*)d_in[3];
    const float* c1b1 = (const float*)d_in[4];
    const float* c1W2 = (const float*)d_in[5];
    const float* c1b2 = (const float*)d_in[6];
    const float* c1g  = (const float*)d_in[7];
    const float* c1be = (const float*)d_in[8];
    const float* c1m  = (const float*)d_in[9];
    const float* c1v  = (const float*)d_in[10];
    const float* cW1  = (const float*)d_in[11];
    const float* cb1  = (const float*)d_in[12];
    const float* cW2  = (const float*)d_in[13];
    const float* cb2  = (const float*)d_in[14];
    const float* cg   = (const float*)d_in[15];
    const float* cbe  = (const float*)d_in[16];
    const float* cm   = (const float*)d_in[17];
    const float* cv   = (const float*)d_in[18];
    const float* l1W  = (const float*)d_in[19];
    const float* l1b  = (const float*)d_in[20];
    const float* l2W  = (const float*)d_in[21];
    const float* l2b  = (const float*)d_in[22];
    float* out = (float*)d_out;

    const int N   = in_sizes[0] / 2;
    const int E   = in_sizes[1] / 2;
    const int G   = out_size;
    const int Lm1 = in_sizes[12] / HID;   // convs_b1 is [L-1, 128]
    const int* src = ei;
    const int* dst = ei + E;

    // -------- workspace layout (f16 activations) --------
    char* wsb = (char*)d_ws;
    const size_t NHb = (size_t)N * HID * sizeof(_Float16);
    _Float16* h16      = (_Float16*)(wsb);
    _Float16* agg16    = (_Float16*)(wsb + NHb);
    _Float16* t16      = (_Float16*)(wsb + 2 * NHb);
    float*    agg2f    = (float*)(wsb + 3 * NHb);
    _Float16* pooled16 = (_Float16*)(wsb + 3 * NHb + (size_t)N * 2 * sizeof(float));
    _Float16* q16      = pooled16 + (size_t)G * HID;

    const int T = 256;
    const int gemmGridN = (N + RPB - 1) / RPB;
    const int gemmGridG = (G + RPB - 1) / RPB;

    // ---- Layer 1: agg over raw [N,2], tiny first linear, WMMA second + BN ----
    gin_zero<<<(int)(((long long)N * 2 + T - 1) / T), T, 0, stream>>>(agg2f, (long long)N * 2);
    gin_agg2<<<(E + T - 1) / T, T, 0, stream>>>(x, src, dst, agg2f, E);
    gin_mlp1<<<(int)(((long long)N * HID + T - 1) / T), T, 0, stream>>>(x, agg2f, c1W1, c1b1, t16, N);
    gin_gemm_single<true><<<gemmGridN, T, 0, stream>>>(t16, c1W2, c1b2,
                                                       c1g, c1be, c1m, c1v, h16, N);

    // ---- Layers 2..L: pk-f16 scatter-add, fused 2-GEMM MLP (in-place on h16) ----
    const long long aggDw = (long long)N * HID / 2;   // f16 buffer in dwords
    for (int i = 0; i < Lm1; ++i) {
        gin_zero<<<(int)((aggDw + T - 1) / T), T, 0, stream>>>((float*)agg16, aggDw);
        gin_agg128<<<(int)(((long long)E * 32 + T - 1) / T), T, 0, stream>>>(h16, src, dst, agg16, E);
        gin_fused_mlp<<<gemmGridN, T, 0, stream>>>(h16, agg16,
                                                   cW1 + (size_t)i * HID * HID, cb1 + (size_t)i * HID,
                                                   cW2 + (size_t)i * HID * HID, cb2 + (size_t)i * HID,
                                                   cg + (size_t)i * HID, cbe + (size_t)i * HID,
                                                   cm + (size_t)i * HID, cv + (size_t)i * HID,
                                                   h16, N);
    }

    // ---- Pool, lin1 (WMMA), lin2 (mat-vec) ----
    const long long poolDw = (long long)G * HID / 2;
    gin_zero<<<(int)((poolDw + T - 1) / T), T, 0, stream>>>((float*)pooled16, poolDw);
    gin_pool<<<(int)(((long long)N * 32 + T - 1) / T), T, 0, stream>>>(h16, bat, pooled16, N);
    gin_gemm_single<false><<<gemmGridG, T, 0, stream>>>(pooled16, l1W, l1b,
                                                        l1b, l1b, l1b, l1b, q16, G);
    gin_lin2<<<(G * 32 + T - 1) / T, T, 0, stream>>>(q16, l2W, l2b, out, G);
}
